// GlobalSelfAttention_32349693673710
// MI455X (gfx1250) — matrix operand order repaired and checked by reading, not compiled, not run on hardware
//
#include <hip/hip_runtime.h>
#include <hip/hip_bf16.h>
#include <math.h>

// ---------------------------------------------------------------------------
// GlobalSelfAttention: B=8, C=256, H=W=64 (N=4096), Cr=32
// bf16 WMMA (16x16x32) flash-attention, fp32 accumulation.
// ---------------------------------------------------------------------------

#define BB 8
#define CC 256
#define NN 4096
#define CR 32

typedef __attribute__((ext_vector_type(16))) __bf16 v16bf;
typedef __attribute__((ext_vector_type(8)))  __bf16 v8bf;
typedef __attribute__((ext_vector_type(8)))  float  v8f;

__device__ __forceinline__ v16bf ld2x8(const __bf16* p0, const __bf16* p1) {
  v8bf a = *(const v8bf*)p0;
  v8bf b = *(const v8bf*)p1;
  v16bf r;
#pragma unroll
  for (int i = 0; i < 8; ++i) { r[i] = a[i]; r[i + 8] = b[i]; }
  return r;
}

__device__ __forceinline__ v8f wmma_bf16(v16bf a, v16bf b, v8f c) {
  return __builtin_amdgcn_wmma_f32_16x16x32_bf16(false, a, false, b,
                                                 (short)0, c, false, false);
}

// ---------------------------------------------------------------------------
// Kernel 1: convert Wt (96x256) and Wo (256x32) to bf16
// ---------------------------------------------------------------------------
__global__ __launch_bounds__(256) void gsa_cvt_weights(
    const float* __restrict__ Wt, const float* __restrict__ Wo,
    __bf16* __restrict__ WtB, __bf16* __restrict__ WoB) {
  int i = blockIdx.x * 256 + threadIdx.x;
  if (i < 96 * CC)  WtB[i] = (__bf16)Wt[i];
  if (i < CC * CR)  WoB[i] = (__bf16)Wo[i];
}

// ---------------------------------------------------------------------------
// Kernel 2: transpose-convert x [B,C,N] f32 -> xT [B,N,C] bf16 (LDS tiled)
// grid: B * (N/32) * (C/32) = 8192 blocks of 256
// ---------------------------------------------------------------------------
__global__ __launch_bounds__(256) void gsa_transpose_x(
    const float* __restrict__ x, __bf16* __restrict__ xT) {
  int idx = blockIdx.x;
  int ct = idx & 7;   idx >>= 3;   // C/32 = 8
  int nt = idx & 127; idx >>= 7;   // N/32 = 128
  int b  = idx;
  __shared__ float tile[32][33];
  int tx = threadIdx.x & 31, ty = threadIdx.x >> 5;
  const float* xp = x + ((size_t)b * CC + ct * 32) * NN + nt * 32;
#pragma unroll
  for (int k = 0; k < 4; ++k)
    tile[ty + k * 8][tx] = xp[(size_t)(ty + k * 8) * NN + tx];
  __syncthreads();
  __bf16* xq = xT + ((size_t)b * NN + nt * 32) * CC + ct * 32;
#pragma unroll
  for (int k = 0; k < 4; ++k)
    xq[(size_t)(ty + k * 8) * CC + tx] = (__bf16)tile[tx][ty + k * 8];
}

// ---------------------------------------------------------------------------
// Kernel 3: QKV projection t = Wt @ x_b  (M=96, K=256, N=4096)
// K,Q stored [B,N,32] bf16 (operand-friendly); V stored [B,32,N] bf16.
// grid: B * (N/128) = 256 blocks, 8 waves; each wave owns 16 j-columns.
// ---------------------------------------------------------------------------
__global__ __launch_bounds__(256) void gsa_qkv(
    const __bf16* __restrict__ WtB, const __bf16* __restrict__ xT,
    __bf16* __restrict__ Kb, __bf16* __restrict__ Qb, __bf16* __restrict__ Vb) {
  const int b    = blockIdx.x >> 5;
  const int jt   = blockIdx.x & 31;
  const int wave = threadIdx.x >> 5;
  const int lane = threadIdx.x & 31;
  const int l16  = lane & 15, h = lane >> 4;
  const int j    = jt * 128 + wave * 16 + l16;

  // Load all eight 32-channel B-operand tiles once (reused for 6 o-tiles).
  const __bf16* xrow = xT + ((size_t)b * NN + j) * CC;
  v16bf bx[8];
#pragma unroll
  for (int kk = 0; kk < 8; ++kk)
    bx[kk] = ld2x8(xrow + kk * 32 + h * 8, xrow + kk * 32 + 16 + h * 8);

#pragma unroll
  for (int ot = 0; ot < 6; ++ot) {
    v8f acc = {};
    const __bf16* wrow = WtB + (size_t)(ot * 16 + l16) * CC;
#pragma unroll
    for (int kk = 0; kk < 8; ++kk) {
      v16bf aw = ld2x8(wrow + kk * 32 + h * 8, wrow + kk * 32 + 16 + h * 8);
      acc = wmma_bf16(aw, bx[kk], acc);
    }
    // D layout: vgpr r, lane: row o = ot*16 + r + h*8, col j = l16
#pragma unroll
    for (int r = 0; r < 8; ++r) {
      int o = ot * 16 + r + h * 8;
      __bf16 val = (__bf16)acc[r];
      if (o < 32)       Kb[((size_t)b * NN + j) * CR + o] = val;
      else if (o < 64)  Qb[((size_t)b * NN + j) * CR + (o - 32)] = val;
      else              Vb[((size_t)b * CR + (o - 64)) * NN + j] = val;
    }
  }
}

// ---------------------------------------------------------------------------
// Kernel 4: fused flash attention + output projection + residual.
// grid: B * (N/128) = 256 blocks, 8 waves; wave owns 16 query columns j.
// scores[i,j] = sum_c K[c,i]Q[c,j]; softmax over i; O[c,j] = sum_i V[c,i]P[i,j]
// out[o,j] = scale * (Wo @ O)[o,j] + x[o,j]
// Software-pipelined: K/V tiles for step n+1 are issued before step n's WMMAs.
// ---------------------------------------------------------------------------
__global__ __launch_bounds__(256) void gsa_attn(
    const __bf16* __restrict__ Kb, const __bf16* __restrict__ Qb,
    const __bf16* __restrict__ Vb, const __bf16* __restrict__ WoB,
    const float* __restrict__ x, const float* __restrict__ scale,
    float* __restrict__ out) {
  const int b    = blockIdx.x >> 5;
  const int jt   = blockIdx.x & 31;
  const int wave = threadIdx.x >> 5;
  const int lane = threadIdx.x & 31;
  const int l16  = lane & 15, h = lane >> 4;
  const int j    = jt * 128 + wave * 16 + l16;

  __shared__ __bf16 Olds[128][40];  // [j_local][c], stride 80B (16B aligned)

  const float s = scale[0];

  // Q operand (B matrix 32x16): lane = col j, K rows = c in [h*8, h*8+8) and [16+h*8, 16+h*8+8)
  const __bf16* qrow = Qb + ((size_t)b * NN + j) * CR;
  v16bf bq = ld2x8(qrow + h * 8, qrow + 16 + h * 8);

  const __bf16* kbase = Kb + ((size_t)b * NN + l16) * CR;   // + i0*CR
  const __bf16* vbase = Vb + ((size_t)b * CR + l16) * NN;   // + i0

  v8f o0 = {}, o1 = {};            // O accumulators: c 0..15 and 16..31
  float m = -INFINITY, l = 0.0f;   // running max / sum for this column j

  // ---- pipeline prologue: tiles for i0 = 0 ----
  v16bf ak0, ak1, av0, av1;
  {
    const __bf16* kr0 = kbase;
    const __bf16* kr1 = kbase + 16 * CR;
    ak0 = ld2x8(kr0 + h * 8, kr0 + 16 + h * 8);
    ak1 = ld2x8(kr1 + h * 8, kr1 + 16 + h * 8);
    const __bf16* vr0 = vbase;
    const __bf16* vr1 = vbase + (size_t)16 * NN;
    av0 = ld2x8(vr0 + h * 16, vr0 + h * 16 + 8);
    av1 = ld2x8(vr1 + h * 16, vr1 + h * 16 + 8);
  }

  for (int i0 = 0; i0 < NN; i0 += 32) {
    // ---- issue next step's tile loads first (wraps to 0 on last step) ----
    int inx = (i0 + 32 < NN) ? (i0 + 32) : 0;
    const __bf16* kr0n = kbase + (size_t)inx * CR;
    const __bf16* kr1n = kr0n + 16 * CR;
    v16bf nk0 = ld2x8(kr0n + h * 8, kr0n + 16 + h * 8);
    v16bf nk1 = ld2x8(kr1n + h * 8, kr1n + 16 + h * 8);
    const __bf16* vr0n = vbase + inx;
    const __bf16* vr1n = vr0n + (size_t)16 * NN;
    v16bf nv0 = ld2x8(vr0n + h * 16, vr0n + h * 16 + 8);
    v16bf nv1 = ld2x8(vr1n + h * 16, vr1n + h * 16 + 8);

    // --- score tiles ---
    v8f zero = {};
    v8f s0 = wmma_bf16(ak0, bq, zero);   // rows i0..i0+15
    v8f s1 = wmma_bf16(ak1, bq, zero);   // rows i0+16..i0+31

    // --- online softmax over the 32 new keys (reduce across half-waves) ---
    float mt = s0[0];
#pragma unroll
    for (int r = 1; r < 8; ++r) mt = fmaxf(mt, s0[r]);
#pragma unroll
    for (int r = 0; r < 8; ++r) mt = fmaxf(mt, s1[r]);
    mt = fmaxf(mt, __shfl_xor(mt, 16, 32));
    float mnew  = fmaxf(m, mt);
    float alpha = __expf(m - mnew);
    m = mnew;
    l *= alpha;
#pragma unroll
    for (int r = 0; r < 8; ++r) { o0[r] *= alpha; o1[r] *= alpha; }

    float p0[8], p1[8], sum = 0.0f;
#pragma unroll
    for (int r = 0; r < 8; ++r) { p0[r] = __expf(s0[r] - m); sum += p0[r]; }
#pragma unroll
    for (int r = 0; r < 8; ++r) { p1[r] = __expf(s1[r] - m); sum += p1[r]; }
    sum += __shfl_xor(sum, 16, 32);
    l += sum;

    // --- transpose P from D layout to B-operand layout (32 i x 16 j) ---
    float c0[8], c1[8];
#pragma unroll
    for (int r = 0; r < 8; ++r) {
      c0[r] = __shfl_xor(p0[r], 16, 32);
      c1[r] = __shfl_xor(p1[r], 16, 32);
    }
    v16bf pb;
#pragma unroll
    for (int v = 0; v < 4; ++v) {           // K rows k = 2v, 2v+1 (k < 8)
      float e0 = h ? c1[2 * v]     : p0[2 * v];
      float e1 = h ? c1[2 * v + 1] : p0[2 * v + 1];
      pb[2 * v] = (__bf16)e0; pb[2 * v + 1] = (__bf16)e1;
    }
#pragma unroll
    for (int v = 4; v < 8; ++v) {           // K rows k = 2v, 2v+1 (k >= 8)
      int r0 = 2 * v - 8;
      float e0 = h ? p1[r0]     : c0[r0];
      float e1 = h ? p1[r0 + 1] : c0[r0 + 1];
      pb[2 * v] = (__bf16)e0; pb[2 * v + 1] = (__bf16)e1;
    }

    // --- O += V * P ---
    o0 = wmma_bf16(av0, pb, o0);
    o1 = wmma_bf16(av1, pb, o1);

    // ---- rotate pipeline registers ----
    ak0 = nk0; ak1 = nk1; av0 = nv0; av1 = nv1;
  }

  // normalize and stage O[c,j] in LDS as [j_local][c] bf16
  float rinv = 1.0f / l;
  int jl = wave * 16 + l16;
#pragma unroll
  for (int r = 0; r < 8; ++r) {
    Olds[jl][r + h * 8]      = (__bf16)(o0[r] * rinv);
    Olds[jl][16 + r + h * 8] = (__bf16)(o1[r] * rinv);
  }
  __syncthreads();

  // --- epilogue: out[o,j] = scale * (Wo @ O)[o,j] + x[o,j] ---
  v16bf bO = ld2x8(&Olds[jl][h * 8], &Olds[jl][16 + h * 8]);
#pragma unroll
  for (int ot = 0; ot < 16; ++ot) {
    const __bf16* wrow = WoB + (size_t)(ot * 16 + l16) * CR;
    v16bf aw = ld2x8(wrow + h * 8, wrow + 16 + h * 8);
    v8f zero = {};
    v8f od = wmma_bf16(aw, bO, zero);
    size_t base = ((size_t)b * CC + ot * 16 + h * 8) * NN + j;
#pragma unroll
    for (int r = 0; r < 8; ++r)
      out[base + (size_t)r * NN] = s * od[r] + x[base + (size_t)r * NN];
  }
}

// ---------------------------------------------------------------------------
// launch
// ---------------------------------------------------------------------------
extern "C" void kernel_launch(void* const* d_in, const int* in_sizes, int n_in,
                              void* d_out, int out_size, void* d_ws, size_t ws_size,
                              hipStream_t stream) {
  const float* x     = (const float*)d_in[0];
  const float* Wt    = (const float*)d_in[1];
  const float* Wo    = (const float*)d_in[2];
  const float* scale = (const float*)d_in[3];
  float* out = (float*)d_out;

  char* ws = (char*)d_ws;
  size_t off = 0;
  __bf16* xT  = (__bf16*)(ws + off); off += (size_t)BB * NN * CC * 2;   // 16 MB
  __bf16* Kb  = (__bf16*)(ws + off); off += (size_t)BB * NN * CR * 2;   // 2 MB
  __bf16* Qb  = (__bf16*)(ws + off); off += (size_t)BB * NN * CR * 2;   // 2 MB
  __bf16* Vb  = (__bf16*)(ws + off); off += (size_t)BB * CR * NN * 2;   // 2 MB
  __bf16* WtB = (__bf16*)(ws + off); off += (size_t)96 * CC * 2;
  __bf16* WoB = (__bf16*)(ws + off); off += (size_t)CC * CR * 2;

  gsa_cvt_weights<<<96, 256, 0, stream>>>(Wt, Wo, WtB, WoB);
  gsa_transpose_x<<<BB * (NN / 32) * (CC / 32), 256, 0, stream>>>(x, xT);
  gsa_qkv<<<BB * (NN / 128), 256, 0, stream>>>(WtB, xT, Kb, Qb, Vb);
  gsa_attn<<<BB * (NN / 128), 256, 0, stream>>>(Kb, Qb, Vb, WoB, x, scale, out);
}
